// SilkNNUE_86466281603145
// MI455X (gfx1250) — matrix operand order repaired and checked by reading, not compiled, not run on hardware
//
#include <hip/hip_runtime.h>

typedef _Float16 v16h __attribute__((ext_vector_type(16)));
typedef _Float16 h8   __attribute__((ext_vector_type(8)));
typedef _Float16 h4   __attribute__((ext_vector_type(4)));
typedef float    v8f  __attribute__((ext_vector_type(8)));
typedef float    f4   __attribute__((ext_vector_type(4)));

#define NIDX   29
#define EMBD   128
#define XSTR   32
#define TILES  4     // 16-sample tiles per wave -> 64 samples/wave, 256/block

// Convert 16 contiguous f32 weights to an f16 B-fragment slice.
__device__ __forceinline__ v16h load_w16(const float* __restrict__ w) {
    const f4* wv = (const f4*)w;
    v16h b;
    #pragma unroll
    for (int q = 0; q < 4; ++q) {
        f4 t = wv[q];
        b[q*4+0] = (_Float16)t[0]; b[q*4+1] = (_Float16)t[1];
        b[q*4+2] = (_Float16)t[2]; b[q*4+3] = (_Float16)t[3];
    }
    return b;
}

// One wave (32 lanes) processes 16 samples per tile, TILES tiles per wave.
// Block = 128 threads = 4 waves.
__global__ __launch_bounds__(128)
void silk_nnue_kernel(const int*   __restrict__ x,
                      const float* __restrict__ emb,
                      const float* __restrict__ W2,
                      const float* __restrict__ b2,
                      const float* __restrict__ W3,
                      const float* __restrict__ b3,
                      const float* __restrict__ W4,
                      float*       __restrict__ out)
{
    __shared__ alignas(16) _Float16 hbuf[4][16][EMBD]; // 16 KB: layer-1 activations (f16)
    __shared__ alignas(16) _Float16 fbuf[4][16][64];   //  8 KB: layer-2 CReLU features (f16)

    const int wave     = threadIdx.x >> 5;
    const int lane     = threadIdx.x & 31;
    const int half     = lane >> 4;     // which 16-lane half of the wave
    const int l15      = lane & 15;
    const int waveBase = blockIdx.x * (64 * TILES) + wave * (16 * TILES);

    // ---- Persistent weight fragments (built once per wave, reused for all tiles) ----
    // B fragment layout: lane holds col N=(tile*16+l15); K = 32*kc + 16*half + {0..15}.
    v16h B2f[4][2];
    #pragma unroll
    for (int kc = 0; kc < 4; ++kc) {
        const int k0 = kc * 32 + half * 16;
        B2f[kc][0] = load_w16(W2 + l15        * 128 + k0);
        B2f[kc][1] = load_w16(W2 + (16 + l15) * 128 + k0);
    }
    v16h B3f[2][2];
    #pragma unroll
    for (int kc = 0; kc < 2; ++kc) {
        const int k0 = kc * 32 + half * 16;
        B3f[kc][0] = load_w16(W3 + l15        * 64 + k0);
        B3f[kc][1] = load_w16(W3 + (16 + l15) * 64 + k0);
    }
    const float bias20 = b2[l15];
    const float bias21 = b2[16 + l15];
    const float bias30 = b3[l15];
    const float bias31 = b3[16 + l15];
    const float w4a  = W4[l15];        // coeff for relu(+z), cols 0..15
    const float w4an = W4[32 + l15];   // coeff for relu(-z), cols 0..15
    const float w4b  = W4[16 + l15];   // cols 16..31
    const float w4bn = W4[48 + l15];

    #pragma unroll 1
    for (int t = 0; t < TILES; ++t) {
        const int rowBase = waveBase + t * 16;

        // ---------- Phase 1: embedding gather-sum + ReLU -> LDS (f16) ----------
        // Whole wave loads one embedding row per index: float4/lane = 512B coalesced.
        {
            const f4* __restrict__ embv = (const f4*)emb;
            #pragma unroll 1
            for (int s = 0; s < 16; ++s) {
                const int  row = rowBase + s;
                const int* xr  = x + row * XSTR;
                f4 acc0 = {0.f, 0.f, 0.f, 0.f};
                f4 acc1 = {0.f, 0.f, 0.f, 0.f};
                #pragma unroll
                for (int j = 0; j < NIDX; j += 2) {
                    const int i0 = xr[j];
                    acc0 += embv[i0 * (EMBD / 4) + lane];
                    if (j + 1 < NIDX) {
                        const int i1 = xr[j + 1];
                        acc1 += embv[i1 * (EMBD / 4) + lane];
                    }
                }
                f4 acc = acc0 + acc1;
                h4 hv;
                hv[0] = (_Float16)fmaxf(acc[0], 0.f);
                hv[1] = (_Float16)fmaxf(acc[1], 0.f);
                hv[2] = (_Float16)fmaxf(acc[2], 0.f);
                hv[3] = (_Float16)fmaxf(acc[3], 0.f);
                *(h4*)&hbuf[wave][s][lane * 4] = hv;   // ds_store_b64
            }
        }
        __syncthreads();

        // ---------- Phase 2: layer 2 GEMM (16x128 @ 128x32) via WMMA ----------
        // A fragment (16-bit A 16x32): lane holds row M=l15;
        // K = 32*kc + 16*half + {0..15}.
        v8f c20 = {}, c21 = {};
        {
            const _Float16* hrow = &hbuf[wave][l15][0];
            #pragma unroll
            for (int kc = 0; kc < 4; ++kc) {
                const int abase = kc * 32 + half * 16;
                h8 lo = *(const h8*)(hrow + abase);        // ds_load_b128
                h8 hi = *(const h8*)(hrow + abase + 8);    // ds_load_b128
                v16h a;
                #pragma unroll
                for (int i = 0; i < 8; ++i) { a[i] = lo[i]; a[8 + i] = hi[i]; }
                c20 = __builtin_amdgcn_wmma_f32_16x16x32_f16(false, a, false, B2f[kc][0], (short)0, c20, false, false);
                c21 = __builtin_amdgcn_wmma_f32_16x16x32_f16(false, a, false, B2f[kc][1], (short)0, c21, false, false);
            }
        }

        // ---------- Phase 3: bias + CReLU -> LDS features (16x64 f16) ----------
        // C layout: lane holds col N=(tile*16+l15); VGPR i holds row M = i + 8*half.
        {
            #pragma unroll
            for (int i = 0; i < 8; ++i) {
                const int m = half * 8 + i;
                const float z0 = c20[i] + bias20;
                const float z1 = c21[i] + bias21;
                fbuf[wave][m][l15]      = (_Float16)fmaxf( z0, 0.f);
                fbuf[wave][m][32 + l15] = (_Float16)fmaxf(-z0, 0.f);
                fbuf[wave][m][16 + l15] = (_Float16)fmaxf( z1, 0.f);
                fbuf[wave][m][48 + l15] = (_Float16)fmaxf(-z1, 0.f);
            }
        }
        __syncthreads();

        // ---------- Phase 4: layer 3 GEMM (16x64 @ 64x32) via WMMA ----------
        v8f c30 = {}, c31 = {};
        {
            const _Float16* frow = &fbuf[wave][l15][0];
            #pragma unroll
            for (int kc = 0; kc < 2; ++kc) {
                const int abase = kc * 32 + half * 16;
                h8 lo = *(const h8*)(frow + abase);
                h8 hi = *(const h8*)(frow + abase + 8);
                v16h a;
                #pragma unroll
                for (int i = 0; i < 8; ++i) { a[i] = lo[i]; a[8 + i] = hi[i]; }
                c30 = __builtin_amdgcn_wmma_f32_16x16x32_f16(false, a, false, B3f[kc][0], (short)0, c30, false, false);
                c31 = __builtin_amdgcn_wmma_f32_16x16x32_f16(false, a, false, B3f[kc][1], (short)0, c31, false, false);
            }
        }

        // ---------- Phase 5: bias + CReLU + dot(W4) + 16-lane reduction ----------
        {
            float p[8];
            #pragma unroll
            for (int i = 0; i < 8; ++i) {
                const float z0 = c30[i] + bias30;
                const float z1 = c31[i] + bias31;
                p[i] = fmaxf( z0, 0.f) * w4a  + fmaxf(-z0, 0.f) * w4an
                     + fmaxf( z1, 0.f) * w4b  + fmaxf(-z1, 0.f) * w4bn;
            }
            // Reduce each p[i] across the 16 lanes of this half-wave.
            #pragma unroll
            for (int i = 0; i < 8; ++i) {
                #pragma unroll
                for (int d = 1; d < 16; d <<= 1)
                    p[i] += __shfl_xor(p[i], d, 16);
            }
            if (l15 < 8) {
                const float v =
                    (l15 == 0) ? p[0] : (l15 == 1) ? p[1] : (l15 == 2) ? p[2] :
                    (l15 == 3) ? p[3] : (l15 == 4) ? p[4] : (l15 == 5) ? p[5] :
                    (l15 == 6) ? p[6] : p[7];
                out[rowBase + half * 8 + l15] = v;
            }
        }
        __syncthreads();   // hbuf/fbuf reuse boundary before next tile
    }
}

extern "C" void kernel_launch(void* const* d_in, const int* in_sizes, int n_in,
                              void* d_out, int out_size, void* d_ws, size_t ws_size,
                              hipStream_t stream) {
    const int*   x   = (const int*)  d_in[0];
    const float* emb = (const float*)d_in[1];
    const float* W2  = (const float*)d_in[2];
    const float* b2  = (const float*)d_in[3];
    const float* W3  = (const float*)d_in[4];
    const float* b3  = (const float*)d_in[5];
    const float* W4  = (const float*)d_in[6];
    float*       out = (float*)d_out;

    const int B = in_sizes[0] / XSTR;              // 131072
    const int blocks = B / (64 * TILES);           // 256 samples per block -> 512 blocks
    silk_nnue_kernel<<<blocks, 128, 0, stream>>>(x, emb, W2, b2, W3, b3, W4, out);
}
